// Attention_44392781971731
// MI455X (gfx1250) — compile-verified, matrix operand order checked by reading, not run on hardware
//
#include <hip/hip_runtime.h>
#include <hip/hip_bf16.h>

// ---------------------------------------------------------------------------
// CDNA5 (gfx1250) attention: bf16 WMMA (f32 accumulate) pipeline.
//   B=4, S=2048, D=1024.  wave32, 8 waves / 256-thread block.
//   Block tile 128x128, wave tile 32x64 -> 8 v_wmma per K-step per wave.
//   Double-buffered LDS staging (one barrier per K-step).
// ---------------------------------------------------------------------------

typedef __bf16 bf16;
typedef __bf16 v16bf __attribute__((ext_vector_type(16)));
typedef __bf16 v8bf  __attribute__((ext_vector_type(8)));
typedef float  v8f   __attribute__((ext_vector_type(8)));

#define LDS_STRIDE 40   // 32 + 8 bf16 pad (80B rows) -> conflict-free b128 access
#define TILE_ELEMS (128 * LDS_STRIDE)
#define SEQ 2048
#define DIM 1024

__device__ __forceinline__ bf16 f2bf(float f) { return (bf16)f; }

// Stage a 128(row) x 32(col) tile into LDS as bf16. 256 threads, 16 elems each.
__device__ __forceinline__ void stage_f32(bf16* lds, const float* __restrict__ src, int ld) {
    int t = threadIdx.x;
    int r = t >> 1;
    int c = (t & 1) * 16;
    const float* g = src + (size_t)r * ld + c;
    float4 a = *(const float4*)g;
    float4 b = *(const float4*)(g + 4);
    float4 e = *(const float4*)(g + 8);
    float4 f = *(const float4*)(g + 12);
    bf16* d = lds + r * LDS_STRIDE + c;
    d[0]  = f2bf(a.x); d[1]  = f2bf(a.y); d[2]  = f2bf(a.z); d[3]  = f2bf(a.w);
    d[4]  = f2bf(b.x); d[5]  = f2bf(b.y); d[6]  = f2bf(b.z); d[7]  = f2bf(b.w);
    d[8]  = f2bf(e.x); d[9]  = f2bf(e.y); d[10] = f2bf(e.z); d[11] = f2bf(e.w);
    d[12] = f2bf(f.x); d[13] = f2bf(f.y); d[14] = f2bf(f.z); d[15] = f2bf(f.w);
}

__device__ __forceinline__ void stage_bf16(bf16* lds, const bf16* __restrict__ src, int ld) {
    int t = threadIdx.x;
    int r = t >> 1;
    int c = (t & 1) * 16;
    const bf16* g = src + (size_t)r * ld + c;
    bf16* d = lds + r * LDS_STRIDE + c;
    *(v8bf*)d       = *(const v8bf*)g;
    *(v8bf*)(d + 8) = *(const v8bf*)(g + 8);
}

// Load one 16x32 bf16 operand fragment (A or row-major B) from an LDS tile.
// ISA layout (16-bit A 16x32): lane L<16 -> row M=L, K in {0..7,16..23};
// lane L+16 -> row M=L, K in {8..15,24..31}.  Two ds_load_b128 per lane.
__device__ __forceinline__ v16bf load_frag(const bf16* tile, int lane) {
    int row  = lane & 15;
    int half = lane >> 4;
    const bf16* p = tile + row * LDS_STRIDE + half * 8;
    v8bf lo = *(const v8bf*)p;          // K = half*8 .. half*8+7
    v8bf hi = *(const v8bf*)(p + 16);   // K = 16+half*8 ..
    v16bf r;
#pragma unroll
    for (int i = 0; i < 8; ++i) { r[i] = lo[i]; r[i + 8] = hi[i]; }
    return r;
}

__device__ __forceinline__ v8f wmma_bf16(v16bf a, v16bf b, v8f c) {
    return __builtin_amdgcn_wmma_f32_16x16x32_bf16(
        /*neg_a=*/false, a, /*neg_b=*/false, b,
        /*c_mod=*/(short)0, c, /*reuse_a=*/false, /*reuse_b=*/false);
}

// Double-buffered 128x128x{K} WMMA core. acc[2][4]: wave tile 32(M) x 64(N).
// SRC_F32: operands are fp32 in global (converted to bf16 while staging).
template <bool SRC_F32, typename SrcT>
__device__ __forceinline__ void gemm_core(bf16* As0, bf16* As1, bf16* Bs0, bf16* Bs1,
                                          const SrcT* __restrict__ Ag,
                                          const SrcT* __restrict__ Bg,
                                          int lda, int ldb, int Ktot,
                                          int wm, int wn, int lane, v8f acc[2][4]) {
    bf16* Ac = As0; bf16* An = As1;
    bf16* Bc = Bs0; bf16* Bn = Bs1;
    if (SRC_F32) {
        stage_f32(Ac, (const float*)Ag, lda);
        stage_f32(Bc, (const float*)Bg, ldb);
    } else {
        stage_bf16(Ac, (const bf16*)Ag, lda);
        stage_bf16(Bc, (const bf16*)Bg, ldb);
    }
    __syncthreads();
    for (int k0 = 0; k0 < Ktot; k0 += 32) {
        if (k0 + 32 < Ktot) {
            if (SRC_F32) {
                stage_f32(An, (const float*)Ag + k0 + 32, lda);
                stage_f32(Bn, (const float*)Bg + k0 + 32, ldb);
            } else {
                stage_bf16(An, (const bf16*)Ag + k0 + 32, lda);
                stage_bf16(Bn, (const bf16*)Bg + k0 + 32, ldb);
            }
            if (k0 + 64 < Ktot)  // global_prefetch_b8: pull tile k+2 toward L2
                __builtin_prefetch((const void*)(Ag + (size_t)(threadIdx.x >> 1) * lda + k0 + 64), 0, 3);
        }
        v16bf a0 = load_frag(Ac + (wm * 32)      * LDS_STRIDE, lane);
        v16bf a1 = load_frag(Ac + (wm * 32 + 16) * LDS_STRIDE, lane);
#pragma unroll
        for (int ni = 0; ni < 4; ++ni) {
            v16bf b = load_frag(Bc + (wn * 64 + ni * 16) * LDS_STRIDE, lane);
            acc[0][ni] = wmma_bf16(a0, b, acc[0][ni]);
            acc[1][ni] = wmma_bf16(a1, b, acc[1][ni]);
        }
        __syncthreads();
        bf16* t;
        t = Ac; Ac = An; An = t;
        t = Bc; Bc = Bn; Bn = t;
    }
}

// ---------------------------------------------------------------------------
// Kernel 1: Y = (x @ W^T + bias) * out_scale   (fp32 in, bf16 out)
// If YT != nullptr, write transposed per-batch: YT[b][n][s]  (for V^T).
// ---------------------------------------------------------------------------
__global__ void __launch_bounds__(256)
qkv_gemm_kernel(const float* __restrict__ X, const float* __restrict__ W,
                const float* __restrict__ bias, bf16* __restrict__ Y,
                bf16* __restrict__ YT, float out_scale, int M, int K, int N) {
    __shared__ bf16 As0[TILE_ELEMS], As1[TILE_ELEMS], Bs0[TILE_ELEMS], Bs1[TILE_ELEMS];
    int tid = threadIdx.x, lane = tid & 31, w = tid >> 5;
    int wm = w & 3, wn = w >> 2;
    int bm = blockIdx.y * 128, bn = blockIdx.x * 128;

    v8f acc[2][4] = {};
    gemm_core<true>(As0, As1, Bs0, Bs1,
                    X + (size_t)bm * K, W + (size_t)bn * K, K, K, K,
                    wm, wn, lane, acc);

    int lc = lane & 15, lh = lane >> 4;
    float bb[4];
#pragma unroll
    for (int ni = 0; ni < 4; ++ni) bb[ni] = bias[bn + wn * 64 + ni * 16 + lc];
#pragma unroll
    for (int mi = 0; mi < 2; ++mi) {
        int mbase = bm + wm * 32 + mi * 16 + lh * 8;
#pragma unroll
        for (int ni = 0; ni < 4; ++ni) {
            int col = bn + wn * 64 + ni * 16 + lc;
#pragma unroll
            for (int v = 0; v < 8; ++v) {
                int m = mbase + v;
                bf16 val = f2bf((acc[mi][ni][v] + bb[ni]) * out_scale);
                if (YT) {
                    int bi = m >> 11, s = m & (SEQ - 1);
                    YT[(size_t)bi * N * SEQ + (size_t)col * SEQ + s] = val;
                } else {
                    Y[(size_t)m * N + col] = val;
                }
            }
        }
    }
}

// ---------------------------------------------------------------------------
// Kernel 2: scores[b][q][k'] = Q[b][q][:] . K[b][k'][:]   (scale folded in Q)
// mask[b][k'] == 0  ->  -3e38 (acts as -inf for softmax).
// ---------------------------------------------------------------------------
__global__ void __launch_bounds__(256)
scores_kernel(const bf16* __restrict__ Q, const bf16* __restrict__ Kc,
              const unsigned char* __restrict__ mask, float* __restrict__ Sc,
              int S, int D) {
    __shared__ bf16 As0[TILE_ELEMS], As1[TILE_ELEMS], Bs0[TILE_ELEMS], Bs1[TILE_ELEMS];
    int tid = threadIdx.x, lane = tid & 31, w = tid >> 5;
    int wm = w & 3, wn = w >> 2;
    int bm = blockIdx.y * 128, bn = blockIdx.x * 128, b = blockIdx.z;
    const bf16* Qb = Q + (size_t)b * S * D;
    const bf16* Kb = Kc + (size_t)b * S * D;
    float* Sb = Sc + (size_t)b * S * S;

    v8f acc[2][4] = {};
    gemm_core<false>(As0, As1, Bs0, Bs1,
                     Qb + (size_t)bm * D, Kb + (size_t)bn * D, D, D, D,
                     wm, wn, lane, acc);

    int lc = lane & 15, lh = lane >> 4;
    float mv[4];
#pragma unroll
    for (int ni = 0; ni < 4; ++ni)
        mv[ni] = mask[(size_t)b * S + bn + wn * 64 + ni * 16 + lc] ? 0.0f : -3.0e38f;
#pragma unroll
    for (int mi = 0; mi < 2; ++mi) {
        int qbase = bm + wm * 32 + mi * 16 + lh * 8;
#pragma unroll
        for (int ni = 0; ni < 4; ++ni) {
            int kc = bn + wn * 64 + ni * 16 + lc;
#pragma unroll
            for (int v = 0; v < 8; ++v) {
                int q = qbase + v;
                Sb[(size_t)q * S + kc] = (mv[ni] == 0.0f) ? acc[mi][ni][v] : mv[ni];
            }
        }
    }
}

// ---------------------------------------------------------------------------
// Kernel 3: row softmax, 1 block per row (2048 elems, 8/thread), bf16 out.
// ---------------------------------------------------------------------------
__global__ void __launch_bounds__(256)
softmax_kernel(const float* __restrict__ Sc, bf16* __restrict__ P, int S) {
    __shared__ float red[256];
    int tid = threadIdx.x;
    size_t row = blockIdx.x;
    const float* src = Sc + row * S;
    bf16* dst = P + row * S;

    float vals[8];
    float lmax = -3.4e38f;
#pragma unroll
    for (int i = 0; i < 8; ++i) {
        vals[i] = src[i * 256 + tid];
        lmax = fmaxf(lmax, vals[i]);
    }
    red[tid] = lmax;
    __syncthreads();
    for (int s = 128; s > 0; s >>= 1) {
        if (tid < s) red[tid] = fmaxf(red[tid], red[tid + s]);
        __syncthreads();
    }
    float rmax = red[0];
    __syncthreads();

    float lsum = 0.0f;
#pragma unroll
    for (int i = 0; i < 8; ++i) {
        vals[i] = __expf(vals[i] - rmax);
        lsum += vals[i];
    }
    red[tid] = lsum;
    __syncthreads();
    for (int s = 128; s > 0; s >>= 1) {
        if (tid < s) red[tid] += red[tid + s];
        __syncthreads();
    }
    float inv = 1.0f / red[0];
#pragma unroll
    for (int i = 0; i < 8; ++i)
        dst[i * 256 + tid] = f2bf(vals[i] * inv);
}

// ---------------------------------------------------------------------------
// Kernel 4: out[b][q][d] = sum_k P[b][q][k] * VT[b][d][k]   (fp32 out)
// ---------------------------------------------------------------------------
__global__ void __launch_bounds__(256)
pv_kernel(const bf16* __restrict__ P, const bf16* __restrict__ VT,
          float* __restrict__ O, int S, int D) {
    __shared__ bf16 As0[TILE_ELEMS], As1[TILE_ELEMS], Bs0[TILE_ELEMS], Bs1[TILE_ELEMS];
    int tid = threadIdx.x, lane = tid & 31, w = tid >> 5;
    int wm = w & 3, wn = w >> 2;
    int bm = blockIdx.y * 128, bn = blockIdx.x * 128, b = blockIdx.z;
    const bf16* Pb = P + (size_t)b * S * S;
    const bf16* Vb = VT + (size_t)b * D * S;
    float* Ob = O + (size_t)b * S * D;

    v8f acc[2][4] = {};
    gemm_core<false>(As0, As1, Bs0, Bs1,
                     Pb + (size_t)bm * S, Vb + (size_t)bn * S, S, S, S,
                     wm, wn, lane, acc);

    int lc = lane & 15, lh = lane >> 4;
#pragma unroll
    for (int mi = 0; mi < 2; ++mi) {
        int qbase = bm + wm * 32 + mi * 16 + lh * 8;
#pragma unroll
        for (int ni = 0; ni < 4; ++ni) {
            int d = bn + wn * 64 + ni * 16 + lc;
#pragma unroll
            for (int v = 0; v < 8; ++v) {
                int q = qbase + v;
                Ob[(size_t)q * D + d] = acc[mi][ni][v];
            }
        }
    }
}

// ---------------------------------------------------------------------------
// Launch: x, mask, Wq, bq, Wk, bk, Wv, bv  ->  d_out [B,S,D] fp32
// ws layout:  [0,16M)  Q bf16          (later overwritten by P, 32 MB)
//             [16,32M) K bf16
//             [32,48M) V^T bf16  [B,D,S]
//             [48,112M) scores f32 [B,S,S]
// ---------------------------------------------------------------------------
extern "C" void kernel_launch(void* const* d_in, const int* in_sizes, int n_in,
                              void* d_out, int out_size, void* d_ws, size_t ws_size,
                              hipStream_t stream) {
    const int B = 4, S = SEQ, D = DIM;
    const int M = B * S;

    const float* x            = (const float*)d_in[0];
    const unsigned char* mask = (const unsigned char*)d_in[1];  // jnp bool = 1 byte
    const float* Wq = (const float*)d_in[2];
    const float* bq = (const float*)d_in[3];
    const float* Wk = (const float*)d_in[4];
    const float* bk = (const float*)d_in[5];
    const float* Wv = (const float*)d_in[6];
    const float* bv = (const float*)d_in[7];

    bf16*  Qb = (bf16*)d_ws;
    bf16*  Kb = Qb + (size_t)M * D;
    bf16*  VT = Kb + (size_t)M * D;
    float* Sc = (float*)(VT + (size_t)M * D);
    bf16*  P  = (bf16*)d_ws;  // overwrites Q+K region after they are consumed

    const float scale = 1.0f / 32.0f;  // 1/sqrt(1024), folded into Q

    dim3 blk(256);
    dim3 gProj(D / 128, M / 128);
    qkv_gemm_kernel<<<gProj, blk, 0, stream>>>(x, Wq, bq, Qb, nullptr, scale, M, D, D);
    qkv_gemm_kernel<<<gProj, blk, 0, stream>>>(x, Wk, bk, Kb, nullptr, 1.0f, M, D, D);
    qkv_gemm_kernel<<<gProj, blk, 0, stream>>>(x, Wv, bv, nullptr, VT, 1.0f, M, D, D);

    dim3 gSc(S / 128, S / 128, B);
    scores_kernel<<<gSc, blk, 0, stream>>>(Qb, Kb, mask, Sc, S, D);

    softmax_kernel<<<dim3(B * S), blk, 0, stream>>>(Sc, P, S);

    dim3 gPV(D / 128, S / 128, B);
    pv_kernel<<<gPV, blk, 0, stream>>>(P, VT, (float*)d_out, S, D);
}
